// Detect_dyhead_8263517077882
// MI455X (gfx1250) — matrix operand order repaired and checked by reading, not compiled, not run on hardware
//
#include <hip/hip_runtime.h>
#include <math.h>

// ---------------------------------------------------------------------------
// Detect_dyhead for MI455X (gfx1250). Feature maps kept in channel-major
// "CN" layout [C][B*H*W]. All convs are bf16 WMMA GEMMs (f32 accumulate)
// with implicit im2col in the LDS staging loop. LDS tiles are stored in
// row-major, k-contiguous order (aligned b128-pair fragment loads), double-buffered,
// and staged with batched unconditional loads (K%32==0, N%64==0 for
// every GEMM in this graph; only the M edge needs clamping).
//
// Assumed d_in order (setup_inputs() insertion order, params traversed in
// nested insertion order): d_in[0..2] = x0,x1,x2; then per level:
//   offset_w, offset_b, dy[0..2]{w,b,gn_g,gn_b}, attn_w, attn_b,
//   fc1_w, fc1_b, fc2_w, fc2_b,
//   cv2{w1,g1,be1,w2,g2,be2,w3,b3}, cv3{w1,g1,be1,w2,g2,be2,w3,b3},
//   [i>0] proj_prev_w, proj_prev_b, [i<2] proj_next_w, proj_next_b
// ---------------------------------------------------------------------------

typedef __bf16 bf16_t;
typedef bf16_t v16bf __attribute__((ext_vector_type(16)));
typedef bf16_t v4bf  __attribute__((ext_vector_type(4)));
typedef float  v8f   __attribute__((ext_vector_type(8)));

#define EPSV 1e-5f

__device__ __forceinline__ float sigf(float x)  { return 1.f / (1.f + __expf(-x)); }
__device__ __forceinline__ float hsigf(float z) { return fminf(fmaxf(z + 3.f, 0.f), 6.f) * (1.f / 6.f); }

// ------------------------------- WMMA GEMM ---------------------------------
// C[M,N] = A[M,K] * B[K,N], f32 in/out, bf16 WMMA accumulate-in-f32.
// MODE 0: B is a dense [K][N] matrix.
// MODE 1: implicit im2col — B is a CN feature [Cin][N] (N = Bn*Hc*Wc),
//         K = Cin*9, 3x3 window, pad 1, stride 1.
//
// Block = 128 threads = 4 waves in a 2x2 grid; block tile 64x64, each wave
// owns 32x32 (2 A-frags x 2 B-frags -> 4 WMMAs per k-step of 32).
// Requires: K % 32 == 0, N % 64 == 0 (true for all GEMMs in this model).
#define GT_M 64
#define GT_N 64
#define GT_K 32

// fragment-order position of k within a 32-wide row (16-bit A layout)
__device__ __forceinline__ int a_frag_pos(int kk) {
  return (kk < 16) ? (((kk >> 3) << 4) + (kk & 7))
                   : ((((kk - 16) >> 3) << 4) + 8 + ((kk - 16) & 7));
}

template <int MODE>
__global__ __launch_bounds__(128)
void k_gemm(const float* __restrict__ A, const float* __restrict__ Bsrc,
            float* __restrict__ C, int M, int N, int K, int Hc, int Wc)
{
  // sA: row-major A tile ([m][k]); sBt: transposed B tile ([n][k])
  __shared__ __align__(32) bf16_t sA [2][GT_M * GT_K];
  __shared__ __align__(32) bf16_t sBt[2][GT_N * GT_K];

  const int tid  = threadIdx.x;
  const int wave = tid >> 5;
  const int lane = tid & 31;
  const int wm   = wave >> 1;       // which 32-row half
  const int wn   = wave & 1;        // which 32-col half
  const int lr   = lane & 15;       // lane row/col within fragment
  const int kh   = lane >> 4;       // k-half selector
  const int m0 = blockIdx.y * GT_M;
  const int n0 = blockIdx.x * GT_N;
  const int hw = Hc * Wc;

  // Staging: issue ALL global loads first (clause-able, one wait), then
  // convert + ds_store. A: 4 float4 tasks/thread. B dense: 4x4 micro-tile.
  // B im2col: 16 scalar gathers with validity bitmask.
  auto stage = [&](int bufi, int k0) {
    bf16_t* dA = sA[bufi];
    bf16_t* dB = sBt[bufi];

    // ---- load phase: A
    float4 va[4];
#pragma unroll
    for (int it = 0; it < 4; ++it) {
      int q = tid + 128 * it;
      int m = q >> 3, t = q & 7;
      int gm = m0 + m;
      int gmc = (gm < M) ? gm : (M - 1);
      va[it] = *(const float4*)(A + (size_t)gmc * K + k0 + 4 * t);
    }

    if (MODE == 0) {
      // ---- load phase: B dense (4x4 k x n micro-tile, coalesced float4)
      int nb = tid >> 3, kb = tid & 7;
      int gn0 = n0 + nb * 4, gk0 = k0 + kb * 4;
      float4 vb[4];
#pragma unroll
      for (int kr = 0; kr < 4; ++kr)
        vb[kr] = *(const float4*)(Bsrc + (size_t)(gk0 + kr) * N + gn0);

      // ---- store phase: A
#pragma unroll
      for (int it = 0; it < 4; ++it) {
        int q = tid + 128 * it;
        int m = q >> 3, t = q & 7;
        float4 v = va[it];
        if (m0 + m >= M) { v.x = 0.f; v.y = 0.f; v.z = 0.f; v.w = 0.f; }
        v4bf o; o[0] = (bf16_t)v.x; o[1] = (bf16_t)v.y; o[2] = (bf16_t)v.z; o[3] = (bf16_t)v.w;
        *(v4bf*)(dA + m * GT_K + 4 * t) = o;   // ds_store_b64
      }
      // ---- store phase: B (transpose: consecutive k contiguous in LDS)
      float r[4][4];
#pragma unroll
      for (int kr = 0; kr < 4; ++kr) {
        r[kr][0] = vb[kr].x; r[kr][1] = vb[kr].y; r[kr][2] = vb[kr].z; r[kr][3] = vb[kr].w;
      }
#pragma unroll
      for (int e = 0; e < 4; ++e) {
        v4bf o; o[0] = (bf16_t)r[0][e]; o[1] = (bf16_t)r[1][e];
        o[2] = (bf16_t)r[2][e]; o[3] = (bf16_t)r[3][e];
        *(v4bf*)(dB + (nb * 4 + e) * GT_K + kb * 4) = o;   // ds_store_b64
      }
    } else {
      // ---- load phase: B im2col gathers (unconditional, clamped)
      float vv[16];
      unsigned okm = 0;
#pragma unroll
      for (int it = 0; it < 16; ++it) {
        int i = tid + 128 * it;
        int r = i >> 6, c = i & 63;     // local k, local n
        int gk = k0 + r, gn = n0 + c;
        int ci = gk / 9, j = gk % 9;
        int b  = gn / hw, p = gn % hw;
        int y  = p / Wc,  x = p % Wc;
        int yy = y + (j / 3) - 1;
        int xx = x + (j % 3) - 1;
        unsigned ok = (yy >= 0) & (yy < Hc) & (xx >= 0) & (xx < Wc);
        int yyc = yy < 0 ? 0 : (yy >= Hc ? Hc - 1 : yy);
        int xxc = xx < 0 ? 0 : (xx >= Wc ? Wc - 1 : xx);
        vv[it] = Bsrc[(size_t)ci * N + (size_t)b * hw + yyc * Wc + xxc];
        okm |= ok << it;
      }
      // ---- store phase: A
#pragma unroll
      for (int it = 0; it < 4; ++it) {
        int q = tid + 128 * it;
        int m = q >> 3, t = q & 7;
        float4 v = va[it];
        if (m0 + m >= M) { v.x = 0.f; v.y = 0.f; v.z = 0.f; v.w = 0.f; }
        v4bf o; o[0] = (bf16_t)v.x; o[1] = (bf16_t)v.y; o[2] = (bf16_t)v.z; o[3] = (bf16_t)v.w;
        *(v4bf*)(dA + m * GT_K + 4 * t) = o;
      }
      // ---- store phase: B
#pragma unroll
      for (int it = 0; it < 16; ++it) {
        int i = tid + 128 * it;
        int r = i >> 6, c = i & 63;
        float v = ((okm >> it) & 1u) ? vv[it] : 0.f;
        dB[c * GT_K + r] = (bf16_t)v;
      }
    }
  };

  v8f acc00 = {}, acc01 = {}, acc10 = {}, acc11 = {};

  stage(0, 0);
  __syncthreads();
  int buf = 0;
  for (int k0 = 0; k0 < K; k0 += GT_K) {
    int nb = buf ^ 1;
    if (k0 + GT_K < K) {
      stage(nb, k0 + GT_K);              // overlap staging with WMMA below
      // prefetch the slab after next (global_prefetch_b8)
      if (k0 + 2 * GT_K < K) {
        int pm = m0 + (tid & (GT_M - 1));
        if (pm < M) __builtin_prefetch(A + (size_t)pm * K + (k0 + 2 * GT_K), 0, 3);
        if (MODE == 0)
          __builtin_prefetch(Bsrc + (size_t)(k0 + 2 * GT_K + (tid & (GT_K - 1))) * N + n0, 0, 3);
      }
    }

    // fragment loads: 32 contiguous, 32B-aligned bytes per lane
    const bf16_t* cA = sA[buf];
    const bf16_t* cB = sBt[buf];
    v16bf a0 = *(const v16bf*)(cA + (wm * 32      + lr) * GT_K + kh * 16);
    v16bf a1 = *(const v16bf*)(cA + (wm * 32 + 16 + lr) * GT_K + kh * 16);
    v16bf b0 = *(const v16bf*)(cB + (wn * 32      + lr) * GT_K + kh * 16);
    v16bf b1 = *(const v16bf*)(cB + (wn * 32 + 16 + lr) * GT_K + kh * 16);

    acc00 = __builtin_amdgcn_wmma_f32_16x16x32_bf16(false, a0, false, b0, (short)0, acc00, false, false);
    acc01 = __builtin_amdgcn_wmma_f32_16x16x32_bf16(false, a0, false, b1, (short)0, acc01, false, false);
    acc10 = __builtin_amdgcn_wmma_f32_16x16x32_bf16(false, a1, false, b0, (short)0, acc10, false, false);
    acc11 = __builtin_amdgcn_wmma_f32_16x16x32_bf16(false, a1, false, b1, (short)0, acc11, false, false);
    __syncthreads();
    buf = nb;
  }

  // store: C/D layout — VGPR r -> M=r (lanes 0-15) / M=r+8 (lanes 16-31)
  auto store_tile = [&](const v8f& a, int mbase, int nbase) {
    int gn = nbase + lr;                 // always < N (N % 64 == 0)
    int mb = mbase + (kh << 3);
#pragma unroll
    for (int r = 0; r < 8; ++r) {
      int gm = mb + r;
      if (gm < M) C[(size_t)gm * N + gn] = a[r];
    }
  };
  store_tile(acc00, m0 + wm * 32,      n0 + wn * 32);
  store_tile(acc01, m0 + wm * 32,      n0 + wn * 32 + 16);
  store_tile(acc10, m0 + wm * 32 + 16, n0 + wn * 32);
  store_tile(acc11, m0 + wm * 32 + 16, n0 + wn * 32 + 16);
}

// --------------------------- pointwise kernels -----------------------------

// NCHW -> CN ([C][B*HW])
__global__ void k_pack_cn(const float* __restrict__ src, float* __restrict__ dst,
                          int C, int HW, int Bn)
{
  int i = blockIdx.x * 256 + threadIdx.x;
  int total = Bn * C * HW;
  if (i >= total) return;
  int b = i / (C * HW);
  int rem = i % (C * HW);
  int c = rem / HW, p = rem % HW;
  dst[(size_t)c * ((size_t)Bn * HW) + (size_t)b * HW + p] = src[i];
}

// C[m][n] += bias[m]; rows >= sig_start get sigmoid (sig_start<0: none)
__global__ void k_bias_rows(float* __restrict__ C, const float* __restrict__ bias,
                            int M, int N, int sig_start)
{
  int i = blockIdx.x * 256 + threadIdx.x;
  if (i >= M * N) return;
  int m = i / N;
  float v = C[i] + bias[m];
  if (sig_start >= 0 && m >= sig_start) v = sigf(v);
  C[i] = v;
}

// Build deformable sampling matrix samp[(c*9+j)][n] from xin CN [C][B*Hi*Wi]
// using offsets/masks in om [27][Nout] (rows 0..17 offsets, 18..26 sigmoid mask)
__global__ void k_deform_sample(const float* __restrict__ xin, const float* __restrict__ om,
                                float* __restrict__ samp, int C, int Hi, int Wi,
                                int Ho, int Wo, int Bn, int stride)
{
  int i = blockIdx.x * 256 + threadIdx.x;
  int Nout = Bn * Ho * Wo;
  int total = C * 9 * Nout;
  if (i >= total) return;
  int n = i % Nout;
  int j = (i / Nout) % 9;
  int c = i / (9 * Nout);
  int hw = Ho * Wo;
  int b = n / hw, p = n % hw;
  int yo = p / Wo, xo = p % Wo;

  float dy = om[(size_t)(2 * j) * Nout + n];
  float dx = om[(size_t)(2 * j + 1) * Nout + n];
  float mk = om[(size_t)(18 + j) * Nout + n];

  float py = (float)(yo * stride) - 1.f + (float)(j / 3) + dy;
  float px = (float)(xo * stride) - 1.f + (float)(j % 3) + dx;
  float y0f = floorf(py), x0f = floorf(px);
  float wy = py - y0f, wx = px - x0f;
  int y0 = (int)y0f, x0 = (int)x0f;

  size_t Nin = (size_t)Bn * Hi * Wi;
  const float* base = xin + (size_t)c * Nin + (size_t)b * Hi * Wi;
  auto g = [&](int yy, int xx) -> float {
    if (yy < 0 || yy >= Hi || xx < 0 || xx >= Wi) return 0.f;
    return base[yy * Wi + xx];
  };
  float v = g(y0, x0)     * (1.f - wy) * (1.f - wx)
          + g(y0, x0 + 1) * (1.f - wy) * wx
          + g(y0 + 1, x0) * wy * (1.f - wx)
          + g(y0 + 1, x0 + 1) * wy * wx;
  samp[i] = v * mk;
}

// GroupNorm over CN layout (in place); grid = B*G blocks
__global__ void k_groupnorm(float* __restrict__ x, const float* __restrict__ gg,
                            const float* __restrict__ bb, int C, int HW, int Bn, int G)
{
  int bid = blockIdx.x;
  int b = bid / G, g = bid % G;
  int cs = C / G;
  size_t N = (size_t)Bn * HW;
  int cnt = cs * HW;
  __shared__ float s1[256], s2[256];
  float su = 0.f, sq = 0.f;
  for (int e = threadIdx.x; e < cnt; e += 256) {
    int ci = e / HW, p = e % HW;
    float v = x[(size_t)(g * cs + ci) * N + (size_t)b * HW + p];
    su += v; sq += v * v;
  }
  s1[threadIdx.x] = su; s2[threadIdx.x] = sq; __syncthreads();
  for (int o = 128; o > 0; o >>= 1) {
    if (threadIdx.x < o) { s1[threadIdx.x] += s1[threadIdx.x + o]; s2[threadIdx.x] += s2[threadIdx.x + o]; }
    __syncthreads();
  }
  float mean = s1[0] / cnt;
  float var  = s2[0] / cnt - mean * mean;
  float inv  = rsqrtf(var + EPSV);
  for (int e = threadIdx.x; e < cnt; e += 256) {
    int ci = e / HW, p = e % HW;
    int ch = g * cs + ci;
    size_t idx = (size_t)ch * N + (size_t)b * HW + p;
    x[idx] = (x[idx] - mean) * inv * gg[ch] + bb[ch];
  }
}

// per-(c,b) spatial mean; grid = C*B blocks; out[c*B+b]
__global__ void k_rowmean(const float* __restrict__ x, float* __restrict__ out,
                          int N, int HW, int Bn)
{
  int bid = blockIdx.x;
  int c = bid / Bn, b = bid % Bn;
  __shared__ float s1[256];
  float su = 0.f;
  for (int p = threadIdx.x; p < HW; p += 256)
    su += x[(size_t)c * N + (size_t)b * HW + p];
  s1[threadIdx.x] = su; __syncthreads();
  for (int o = 128; o > 0; o >>= 1) {
    if (threadIdx.x < o) s1[threadIdx.x] += s1[threadIdx.x + o];
    __syncthreads();
  }
  if (threadIdx.x == 0) out[bid] = s1[0] / HW;
}

// attention scalar per (k,b): hsig(relu(sum_c gap*aw + ab))
__global__ void k_attn(const float* __restrict__ gap, const float* __restrict__ aw,
                       const float* __restrict__ ab, float* __restrict__ out,
                       int C, int Bn, int Kf)
{
  int bid = blockIdx.x; // k*B + b
  int k = bid / Bn, b = bid % Bn;
  __shared__ float s1[256];
  float su = 0.f;
  for (int c = threadIdx.x; c < C; c += 256)
    su += gap[(size_t)k * C * Bn + (size_t)c * Bn + b] * aw[c];
  s1[threadIdx.x] = su; __syncthreads();
  for (int o = 128; o > 0; o >>= 1) {
    if (threadIdx.x < o) s1[threadIdx.x] += s1[threadIdx.x + o];
    __syncthreads();
  }
  if (threadIdx.x == 0) out[bid] = hsigf(fmaxf(s1[0] + ab[0], 0.f));
}

// out[c][n] = mean_k feats[k][c][n] * attn[k][b]
__global__ void k_combine(const float* __restrict__ feats, const float* __restrict__ attn,
                          float* __restrict__ out, int C, int N, int HW, int Bn,
                          int Kf, int fstride)
{
  int i = blockIdx.x * 256 + threadIdx.x;
  if (i >= C * N) return;
  int n = i % N, b = n / HW;
  float s = 0.f;
  for (int k = 0; k < Kf; ++k) s += feats[(size_t)k * fstride + i] * attn[k * Bn + b];
  out[i] = s / (float)Kf;
}

// tiny dense layers for DyReLU
__global__ void k_fc1(const float* __restrict__ y, const float* __restrict__ w,
                      const float* __restrict__ bia, float* __restrict__ z,
                      int C, int C4, int Bn)
{
  int t = blockIdx.x * 256 + threadIdx.x;
  if (t >= Bn * C4) return;
  int b = t / C4, h = t % C4;
  float s = bia[h];
  for (int c = 0; c < C; ++c) s += y[(size_t)c * Bn + b] * w[(size_t)h * C + c];
  z[t] = fmaxf(s, 0.f);
}

__global__ void k_fc2(const float* __restrict__ z1, const float* __restrict__ w,
                      const float* __restrict__ bia, float* __restrict__ z2,
                      int C4, int O, int Bn)
{
  int t = blockIdx.x * 256 + threadIdx.x;
  if (t >= Bn * O) return;
  int b = t / O, o = t % O;
  float s = bia[o];
  for (int h = 0; h < C4; ++h) s += z1[(size_t)b * C4 + h] * w[(size_t)o * C4 + h];
  z2[t] = hsigf(s);
}

__global__ void k_dyrelu_apply(float* __restrict__ x, const float* __restrict__ z2,
                               int C, int N, int HW)
{
  int i = blockIdx.x * 256 + threadIdx.x;
  if (i >= C * N) return;
  int c = i / N, n = i % N, b = n / HW;
  const float* zb = z2 + (size_t)b * 4 * C;
  float a1 = (zb[c] - 0.5f) * 2.f + 1.f;
  float b1 = zb[C + c] - 0.5f;
  float a2 = (zb[2 * C + c] - 0.5f) * 2.f;
  float b2 = zb[3 * C + c] - 0.5f;
  float v = x[i];
  x[i] = fmaxf(v * a1 + b1, v * a2 + b2);
}

// training-mode BatchNorm over a row (channel) of [M][N] + SiLU; grid = M
__global__ void k_bn_silu(float* __restrict__ x, const float* __restrict__ gg,
                          const float* __restrict__ be, int N)
{
  int m = blockIdx.x;
  float* row = x + (size_t)m * N;
  __shared__ float s1[256], s2[256];
  float su = 0.f, sq = 0.f;
  for (int n = threadIdx.x; n < N; n += 256) { float v = row[n]; su += v; sq += v * v; }
  s1[threadIdx.x] = su; s2[threadIdx.x] = sq; __syncthreads();
  for (int o = 128; o > 0; o >>= 1) {
    if (threadIdx.x < o) { s1[threadIdx.x] += s1[threadIdx.x + o]; s2[threadIdx.x] += s2[threadIdx.x + o]; }
    __syncthreads();
  }
  float mean = s1[0] / N;
  float var  = s2[0] / N - mean * mean;
  float scale = gg[m] * rsqrtf(var + EPSV);
  float shift = be[m] - mean * scale;
  for (int n = threadIdx.x; n < N; n += 256) {
    float v = row[n] * scale + shift;
    row[n] = v * sigf(v);
  }
}

// src[Co][N] + bias -> d_out level block [B][144][H][W] at channel offset
__global__ void k_final_store(const float* __restrict__ src, const float* __restrict__ bias,
                              float* __restrict__ out, int Co, int choff, int N, int HW)
{
  int i = blockIdx.x * 256 + threadIdx.x;
  if (i >= Co * N) return;
  int m = i / N, n = i % N;
  int b = n / HW, p = n % HW;
  out[((size_t)b * 144 + choff + m) * HW + p] = src[i] + bias[m];
}

// ------------------------------- host side ---------------------------------

struct Lvl {
  const float *offset_w, *offset_b;
  const float *dy_w[3], *dy_b[3], *dy_gng[3], *dy_gnb[3];
  const float *attn_w, *attn_b;
  const float *fc1_w, *fc1_b, *fc2_w, *fc2_b;
  const float *cv2_w1, *cv2_g1, *cv2_be1, *cv2_w2, *cv2_g2, *cv2_be2, *cv2_w3, *cv2_b3;
  const float *cv3_w1, *cv3_g1, *cv3_be1, *cv3_w2, *cv3_g2, *cv3_be2, *cv3_w3, *cv3_b3;
  const float *proj_prev_w, *proj_prev_b, *proj_next_w, *proj_next_b;
};

static inline dim3 ew(long long total) { return dim3((unsigned)((total + 255) / 256)); }

extern "C" void kernel_launch(void* const* d_in, const int* in_sizes, int n_in,
                              void* d_out, int out_size, void* d_ws, size_t ws_size,
                              hipStream_t stream)
{
  (void)in_sizes; (void)n_in; (void)out_size; (void)ws_size;

  const float* xin[3] = {(const float*)d_in[0], (const float*)d_in[1], (const float*)d_in[2]};
  auto P = [&](int i) { return (const float*)d_in[i]; };

  Lvl L[3];
  int idx = 3;
  for (int i = 0; i < 3; ++i) {
    Lvl& l = L[i];
    l.offset_w = P(idx++); l.offset_b = P(idx++);
    for (int k = 0; k < 3; ++k) {
      l.dy_w[k] = P(idx++); l.dy_b[k] = P(idx++);
      l.dy_gng[k] = P(idx++); l.dy_gnb[k] = P(idx++);
    }
    l.attn_w = P(idx++); l.attn_b = P(idx++);
    l.fc1_w = P(idx++); l.fc1_b = P(idx++);
    l.fc2_w = P(idx++); l.fc2_b = P(idx++);
    l.cv2_w1 = P(idx++); l.cv2_g1 = P(idx++); l.cv2_be1 = P(idx++);
    l.cv2_w2 = P(idx++); l.cv2_g2 = P(idx++); l.cv2_be2 = P(idx++);
    l.cv2_w3 = P(idx++); l.cv2_b3 = P(idx++);
    l.cv3_w1 = P(idx++); l.cv3_g1 = P(idx++); l.cv3_be1 = P(idx++);
    l.cv3_w2 = P(idx++); l.cv3_g2 = P(idx++); l.cv3_be2 = P(idx++);
    l.cv3_w3 = P(idx++); l.cv3_b3 = P(idx++);
    l.proj_prev_w = l.proj_prev_b = l.proj_next_w = l.proj_next_b = nullptr;
    if (i > 0) { l.proj_prev_w = P(idx++); l.proj_prev_b = P(idx++); }
    if (i < 2) { l.proj_next_w = P(idx++); l.proj_next_b = P(idx++); }
  }

  const int CHs[3] = {256, 512, 1024};
  const int Hs[3]  = {64, 32, 16};
  const int Bn = 2;
  const size_t OUTOFF[3] = {0, (size_t)2 * 144 * 64 * 64,
                            (size_t)2 * 144 * 64 * 64 + (size_t)2 * 144 * 32 * 32};

  // workspace carve (floats)
  float* wsf = (float*)d_ws;
  size_t cur = 0;
  auto alloc = [&](size_t n) { float* p = wsf + cur; cur += n; return p; };
  const int FSTRIDE = 2097152;
  float* XCN[3];
  XCN[0] = alloc(2097152); XCN[1] = alloc(1048576); XCN[2] = alloc(524288);
  float* PPREV = alloc(4194304);
  float* PNEXT = alloc(524288);
  float* BIG   = alloc(18874368);   // deformable sampling matrix
  float* OM    = alloc(221184);
  float* FEATS = alloc((size_t)3 * FSTRIDE);
  float* FEA   = alloc(2097152);
  float* H1    = alloc(2097152);
  float* H2    = alloc(2097152);
  float* HOUT  = alloc(655360);
  float* GAP   = alloc(8192);
  float* ATTN  = alloc(16);
  float* Y     = alloc(2048);
  float* Z1    = alloc(512);
  float* Z2    = alloc(8192);

  auto gemm = [&](const float* A, const float* Bm, float* Cc, int M, int Nn, int K,
                  int mode, int Hc, int Wc) {
    dim3 g(Nn / GT_N, (M + GT_M - 1) / GT_M);   // N % 64 == 0 always
    if (mode == 0) k_gemm<0><<<g, 128, 0, stream>>>(A, Bm, Cc, M, Nn, K, Hc, Wc);
    else           k_gemm<1><<<g, 128, 0, stream>>>(A, Bm, Cc, M, Nn, K, Hc, Wc);
  };

  // pack inputs to CN
  for (int i = 0; i < 3; ++i) {
    int HW = Hs[i] * Hs[i];
    k_pack_cn<<<ew((long long)Bn * CHs[i] * HW), 256, 0, stream>>>(xin[i], XCN[i], CHs[i], HW, Bn);
  }

  for (int i = 0; i < 3; ++i) {
    const Lvl& l = L[i];
    const int C = CHs[i], H = Hs[i], W = H, HW = H * W, N = Bn * HW;
    float* curf = XCN[i];

    // 1x1 projections of neighbor levels (at their own resolution)
    float* prevp = nullptr; int Hprev = 0;
    if (i > 0) {
      Hprev = Hs[i - 1];
      int Np = Bn * Hprev * Hprev;
      gemm(l.proj_prev_w, XCN[i - 1], PPREV, C, Np, CHs[i - 1], 0, 0, 0);
      k_bias_rows<<<ew((long long)C * Np), 256, 0, stream>>>(PPREV, l.proj_prev_b, C, Np, -1);
      prevp = PPREV;
    }
    float* nxtp = nullptr; int Hnxt = 0;
    if (i < 2) {
      Hnxt = Hs[i + 1];
      int Nn = Bn * Hnxt * Hnxt;
      gemm(l.proj_next_w, XCN[i + 1], PNEXT, C, Nn, CHs[i + 1], 0, 0, 0);
      k_bias_rows<<<ew((long long)C * Nn), 256, 0, stream>>>(PNEXT, l.proj_next_b, C, Nn, -1);
      nxtp = PNEXT;
    }

    // offset conv (3x3 implicit im2col) -> om [27][N]; sigmoid on mask rows
    gemm(l.offset_w, curf, OM, 27, N, C * 9, 1, H, W);
    k_bias_rows<<<ew((long long)27 * N), 256, 0, stream>>>(OM, l.offset_b, 27, N, 18);

    // deformable branches
    int Kf = 0;
    auto branch = [&](const float* src, int Hi, int Wi, int stride, int dyi) {
      k_deform_sample<<<ew((long long)C * 9 * N), 256, 0, stream>>>(
          src, OM, BIG, C, Hi, Wi, H, W, Bn, stride);
      float* ft = FEATS + (size_t)Kf * FSTRIDE;
      gemm(l.dy_w[dyi], BIG, ft, C, N, C * 9, 0, 0, 0);
      k_bias_rows<<<ew((long long)C * N), 256, 0, stream>>>(ft, l.dy_b[dyi], C, N, -1);
      k_groupnorm<<<Bn * 16, 256, 0, stream>>>(ft, l.dy_gng[dyi], l.dy_gnb[dyi], C, HW, Bn, 16);
      k_rowmean<<<C * Bn, 256, 0, stream>>>(ft, GAP + (size_t)Kf * C * Bn, N, HW, Bn);
      ++Kf;
    };
    branch(curf, H, W, 1, 1);                       // current level, dy[1]
    if (prevp) branch(prevp, Hprev, Hprev, 2, 2);   // higher-res prev, dy[2], stride 2
    if (nxtp)  branch(nxtp,  Hnxt,  Hnxt,  1, 0);   // lower-res next, dy[0]

    // scale attention + combine
    k_attn<<<Kf * Bn, 256, 0, stream>>>(GAP, l.attn_w, l.attn_b, ATTN, C, Bn, Kf);
    k_combine<<<ew((long long)C * N), 256, 0, stream>>>(FEATS, ATTN, FEA, C, N, HW, Bn, Kf, FSTRIDE);

    // DyReLU
    int C4 = C / 4;
    k_rowmean<<<C * Bn, 256, 0, stream>>>(FEA, Y, N, HW, Bn);
    k_fc1<<<ew((long long)Bn * C4), 256, 0, stream>>>(Y, l.fc1_w, l.fc1_b, Z1, C, C4, Bn);
    k_fc2<<<ew((long long)Bn * 4 * C), 256, 0, stream>>>(Z1, l.fc2_w, l.fc2_b, Z2, C4, 4 * C, Bn);
    k_dyrelu_apply<<<ew((long long)C * N), 256, 0, stream>>>(FEA, Z2, C, N, HW);

    // heads: conv3x3+BN+SiLU, conv3x3+BN+SiLU, conv1x1+bias
    auto run_head = [&](const float* w1, const float* g1, const float* be1,
                        const float* w2, const float* g2, const float* be2,
                        const float* w3, const float* b3, int Cmid, int Co, int choff) {
      gemm(w1, FEA, H1, Cmid, N, C * 9, 1, H, W);
      k_bn_silu<<<Cmid, 256, 0, stream>>>(H1, g1, be1, N);
      gemm(w2, H1, H2, Cmid, N, Cmid * 9, 1, H, W);
      k_bn_silu<<<Cmid, 256, 0, stream>>>(H2, g2, be2, N);
      gemm(w3, H2, HOUT, Co, N, Cmid, 0, 0, 0);
      k_final_store<<<ew((long long)Co * N), 256, 0, stream>>>(
          HOUT, b3, (float*)d_out + OUTOFF[i], Co, choff, N, HW);
    };
    run_head(l.cv2_w1, l.cv2_g1, l.cv2_be1, l.cv2_w2, l.cv2_g2, l.cv2_be2,
             l.cv2_w3, l.cv2_b3, 64, 64, 0);
    run_head(l.cv3_w1, l.cv3_g1, l.cv3_be1, l.cv3_w2, l.cv3_g2, l.cv3_be2,
             l.cv3_w3, l.cv3_b3, 256, 80, 64);
  }
}
